// LocationSlayerRandom_6408091205839
// MI455X (gfx1250) — matrix operand order repaired and checked by reading, not compiled, not run on hardware
//
#include <hip/hip_runtime.h>

typedef __attribute__((ext_vector_type(16))) _Float16 v16h;
typedef __attribute__((ext_vector_type(8)))  _Float16 v8h;
typedef __attribute__((ext_vector_type(8)))  float    v8f;

namespace {
constexpr int   Bn    = 32;
constexpr int   CIN   = 156;
constexpr int   CINP  = 160;   // K padded to multiple of 32
constexpr int   Tn    = 2048;
constexpr int   HIDN  = 512;
constexpr int   OUTC  = 20;
constexpr int   TOUT  = 2204;  // 2048 + 156
constexpr float THETA = 10.0f;
constexpr float ALPHA = 0.90483741803595957f; // exp(-1/10)
}

// ---------------- WMMA fragment helpers (CDNA5 16x16x32 f16 layouts) --------
__device__ __forceinline__ v16h frag_cat(v8h lo, v8h hi) {
  return __builtin_shufflevector(lo, hi, 0,1,2,3,4,5,6,7,8,9,10,11,12,13,14,15);
}

// A: 16(M) x 32(K), row-major in memory (lda halves per row).
// lane l<16 : row l,  K halves {k0+0..7,  k0+16..23}
// lane l>=16: row l-16, K halves {k0+8..15, k0+24..31}
__device__ __forceinline__ v16h load_frag_a(const _Float16* __restrict__ A,
                                            int lda, int lane, int k0) {
  const _Float16* p = A + (lane & 15) * lda + k0 + ((lane >> 4) << 3);
  v8h lo = *(const v8h*)(p);
  v8h hi = *(const v8h*)(p + 16);
  return frag_cat(lo, hi);
}

// B: 32(K) x 16(N), stored with each column's K-run contiguous (ldb halves/col).
// lane l: column (n0 + (l&15)), K halves {k0+8h+0..7, k0+16+8h+0..7}, h=(l>=16)
__device__ __forceinline__ v16h load_frag_b(const _Float16* __restrict__ B,
                                            int ldb, int n0, int lane, int k0) {
  const _Float16* p = B + (n0 + (lane & 15)) * ldb + k0 + ((lane >> 4) << 3);
  v8h lo = *(const v8h*)(p);
  v8h hi = *(const v8h*)(p + 16);
  return frag_cat(lo, hi);
}

__device__ __forceinline__ v8f wmma_f16(v16h a, v16h b, v8f c) {
  return __builtin_amdgcn_wmma_f32_16x16x32_f16(false, a, false, b, (short)0, c,
                                                false, false);
}

// ---------------- prep: f32 -> f16 with zero padding -------------------------
__global__ void k_cvt_pad(const float* __restrict__ src, _Float16* __restrict__ dst,
                          int rows, int cols, int ldo, int total) {
  int i = blockIdx.x * blockDim.x + threadIdx.x;
  if (i >= total) return;
  int r = i / ldo, c = i - r * ldo;
  float v = (r < rows && c < cols) ? src[r * cols + c] : 0.0f;
  dst[i] = (_Float16)v;
}

// Xt[b][t][c160] = X[b][c][t]  (K-contiguous B operand for GEMM1)
__global__ void k_build_xt(const float* __restrict__ X, _Float16* __restrict__ Xt) {
  int i = blockIdx.x * blockDim.x + threadIdx.x;   // over Bn*Tn*CINP
  int c  = i % CINP;
  int bt = i / CINP;
  int t  = bt & (Tn - 1);
  int b  = bt >> 11;
  float v = (c < CIN) ? X[(b * CIN + c) * Tn + t] : 0.0f;
  Xt[i] = (_Float16)v;
}

// Xp[b][c][t] = X[b][perm[c]][t]  (K=t contiguous B operand for GEMM3)
__global__ void k_build_xp(const float* __restrict__ X, const int* __restrict__ perm,
                           _Float16* __restrict__ Xp) {
  int i = blockIdx.x * blockDim.x + threadIdx.x;   // over Bn*CIN*Tn
  int t  = i & (Tn - 1);
  int bc = i >> 11;
  int c  = bc % CIN;
  int b  = bc / CIN;
  Xp[i] = (_Float16)X[(b * CIN + perm[c]) * Tn + t];
}

// ---------------- GEMM1: Y1h[b][o][t] = W1h(512x160) * Xt ------------------
__global__ __launch_bounds__(256) void k_gemm1(const _Float16* __restrict__ A,
                                               const _Float16* __restrict__ Bm,
                                               _Float16* __restrict__ Y1) {
  const int lane = threadIdx.x & 31;
  const int m0   = (threadIdx.x >> 5) << 6;   // 4 M-tiles per wave
  const int n0   = blockIdx.x << 4;           // 4096 N-tiles
  v8f acc[4] = {};
#pragma unroll
  for (int k = 0; k < CINP; k += 32) {
    v16h bf = load_frag_b(Bm, CINP, n0, lane, k);
#pragma unroll
    for (int m = 0; m < 4; ++m) {
      v16h af = load_frag_a(A + (m0 + (m << 4)) * CINP, CINP, lane, k);
      acc[m] = wmma_f16(af, bf, acc[m]);
    }
  }
  const int n = n0 + (lane & 15);
  const int b = n >> 11, t = n & (Tn - 1);
  const int rb = (lane >> 4) << 3;
#pragma unroll
  for (int m = 0; m < 4; ++m)
#pragma unroll
    for (int r = 0; r < 8; ++r) {
      int M = m0 + (m << 4) + rb + r;
      Y1[(b * HIDN + M) * Tn + t] = (_Float16)acc[m][r];
    }
}

// ---------------- scan1: psp+spike over t; S1h[b][t][h] ---------------------
__global__ void k_scan1(const _Float16* __restrict__ Y1, _Float16* __restrict__ S1) {
  int i = blockIdx.x * blockDim.x + threadIdx.x;   // b*HIDN + h
  if (i >= Bn * HIDN) return;
  int b = i / HIDN, h = i - b * HIDN;
  const _Float16* src = Y1 + i * Tn;
  float c = 0.0f;
  for (int t = 0; t < Tn; ++t) {
    c = ALPHA * c + (float)src[t];
    S1[(b * Tn + t) * HIDN + h] = (c >= THETA) ? (_Float16)1.0f : (_Float16)0.0f;
  }
}

// ---------------- GEMM2: Y2[b][o20][t] = W2h(32x512) * S1h ------------------
__global__ __launch_bounds__(256) void k_gemm2(const _Float16* __restrict__ A,
                                               const _Float16* __restrict__ Bm,
                                               float* __restrict__ Y2) {
  const int lane = threadIdx.x & 31;
  const int n0 = ((blockIdx.x << 3) + (threadIdx.x >> 5)) << 4;
  v8f acc0 = {}, acc1 = {};
#pragma unroll 4
  for (int k = 0; k < HIDN; k += 32) {
    v16h bf = load_frag_b(Bm, HIDN, n0, lane, k);
    acc0 = wmma_f16(load_frag_a(A, HIDN, lane, k), bf, acc0);
    acc1 = wmma_f16(load_frag_a(A + 16 * HIDN, HIDN, lane, k), bf, acc1);
  }
  const int n = n0 + (lane & 15);
  const int b = n >> 11, t = n & (Tn - 1);
  const int rb = (lane >> 4) << 3;
#pragma unroll
  for (int r = 0; r < 8; ++r) {
    int M0 = rb + r;
    if (M0 < OUTC) Y2[(b * OUTC + M0) * Tn + t] = acc0[r];
    int M1 = 16 + rb + r;
    if (M1 < OUTC) Y2[(b * OUTC + M1) * Tn + t] = acc1[r];
  }
}

// ---------------- scan2: psp+spike over t -> out[:, :, 0:2048] --------------
__global__ void k_scan2(const float* __restrict__ Y2, float* __restrict__ out) {
  int i = blockIdx.x * blockDim.x + threadIdx.x;   // b*OUTC + o
  if (i >= Bn * OUTC) return;
  const float* src = Y2 + i * Tn;
  float* dst = out + (long)i * TOUT;
  float c = 0.0f;
  for (int t = 0; t < Tn; ++t) {
    c = ALPHA * c + src[t];
    dst[t] = (c >= THETA) ? 1.0f : 0.0f;
  }
}

// ---------------- GEMM3: Z1[b][o][c] = Wl1h(512x2048) * Xp ------------------
__global__ __launch_bounds__(256) void k_gemm3(const _Float16* __restrict__ A,
                                               const _Float16* __restrict__ Bm,
                                               float* __restrict__ Z1) {
  const int lane = threadIdx.x & 31;
  const int m0   = (threadIdx.x >> 5) << 6;
  const int n0   = blockIdx.x << 4;           // 312 N-tiles
  v8f acc[4] = {};
#pragma unroll 2
  for (int k = 0; k < Tn; k += 32) {
    v16h bf = load_frag_b(Bm, Tn, n0, lane, k);
#pragma unroll
    for (int m = 0; m < 4; ++m) {
      v16h af = load_frag_a(A + (m0 + (m << 4)) * Tn, Tn, lane, k);
      acc[m] = wmma_f16(af, bf, acc[m]);
    }
  }
  const int n = n0 + (lane & 15);
  const int b = n / CIN, c = n - b * CIN;
  const int rb = (lane >> 4) << 3;
#pragma unroll
  for (int m = 0; m < 4; ++m)
#pragma unroll
    for (int r = 0; r < 8; ++r) {
      int M = m0 + (m << 4) + rb + r;
      Z1[(b * HIDN + M) * CIN + c] = acc[m][r];
    }
}

// ---------------- scan3: psp+spike over c; L1h[b][c][h] ---------------------
__global__ void k_scan3(const float* __restrict__ Z1, _Float16* __restrict__ L1) {
  int i = blockIdx.x * blockDim.x + threadIdx.x;   // b*HIDN + h
  if (i >= Bn * HIDN) return;
  int b = i / HIDN, h = i - b * HIDN;
  const float* src = Z1 + i * CIN;
  float c = 0.0f;
  for (int cc = 0; cc < CIN; ++cc) {
    c = ALPHA * c + src[cc];
    L1[(b * CIN + cc) * HIDN + h] = (c >= THETA) ? (_Float16)1.0f : (_Float16)0.0f;
  }
}

// ---------------- GEMM4: Z2[b][o20][c] = Wl2h(32x512) * L1h -----------------
__global__ __launch_bounds__(256) void k_gemm4(const _Float16* __restrict__ A,
                                               const _Float16* __restrict__ Bm,
                                               float* __restrict__ Z2) {
  const int lane = threadIdx.x & 31;
  const int n0 = ((blockIdx.x << 3) + (threadIdx.x >> 5)) << 4;
  v8f acc0 = {}, acc1 = {};
#pragma unroll 4
  for (int k = 0; k < HIDN; k += 32) {
    v16h bf = load_frag_b(Bm, HIDN, n0, lane, k);
    acc0 = wmma_f16(load_frag_a(A, HIDN, lane, k), bf, acc0);
    acc1 = wmma_f16(load_frag_a(A + 16 * HIDN, HIDN, lane, k), bf, acc1);
  }
  const int n = n0 + (lane & 15);
  const int b = n / CIN, c = n - b * CIN;
  const int rb = (lane >> 4) << 3;
#pragma unroll
  for (int r = 0; r < 8; ++r) {
    int M0 = rb + r;
    if (M0 < OUTC) Z2[(b * OUTC + M0) * CIN + c] = acc0[r];
    int M1 = 16 + rb + r;
    if (M1 < OUTC) Z2[(b * OUTC + M1) * CIN + c] = acc1[r];
  }
}

// ---------------- scan4: psp+spike over c -> out[:, :, 2048:2204] -----------
__global__ void k_scan4(const float* __restrict__ Z2, float* __restrict__ out) {
  int i = blockIdx.x * blockDim.x + threadIdx.x;   // b*OUTC + o
  if (i >= Bn * OUTC) return;
  const float* src = Z2 + i * CIN;
  float* dst = out + (long)i * TOUT + Tn;
  float c = 0.0f;
  for (int cc = 0; cc < CIN; ++cc) {
    c = ALPHA * c + src[cc];
    dst[cc] = (c >= THETA) ? 1.0f : 0.0f;
  }
}

// ---------------- host launcher ---------------------------------------------
extern "C" void kernel_launch(void* const* d_in, const int* in_sizes, int n_in,
                              void* d_out, int out_size, void* d_ws, size_t ws_size,
                              hipStream_t stream) {
  const float* X    = (const float*)d_in[0];  // [32,156,1,1,2048]
  const float* W1   = (const float*)d_in[1];  // [512,156]
  const float* W2   = (const float*)d_in[2];  // [20,512]
  const float* Wl1  = (const float*)d_in[3];  // [512,2048]
  const float* Wl2  = (const float*)d_in[4];  // [20,512]
  const int*   perm = (const int*)d_in[5];    // [156]
  float* out = (float*)d_out;                 // [32,20,2204]

  char* ws = (char*)d_ws;
  size_t off = 0;
  auto take = [&](size_t bytes) -> char* {
    char* p = ws + off;
    off = (off + bytes + 255) & ~(size_t)255;
    return p;
  };
  _Float16* W1h  = (_Float16*)take((size_t)HIDN * CINP * 2);
  _Float16* W2h  = (_Float16*)take((size_t)32 * HIDN * 2);
  _Float16* Wl1h = (_Float16*)take((size_t)HIDN * Tn * 2);
  _Float16* Wl2h = (_Float16*)take((size_t)32 * HIDN * 2);
  _Float16* Xt   = (_Float16*)take((size_t)Bn * Tn * CINP * 2);
  _Float16* Xp   = (_Float16*)take((size_t)Bn * CIN * Tn * 2);
  _Float16* Y1h  = (_Float16*)take((size_t)Bn * HIDN * Tn * 2);
  _Float16* S1h  = (_Float16*)take((size_t)Bn * Tn * HIDN * 2);
  float*    Y2   = (float*)take((size_t)Bn * OUTC * Tn * 4);
  float*    Z1   = (float*)take((size_t)Bn * HIDN * CIN * 4);
  _Float16* L1h  = (_Float16*)take((size_t)Bn * CIN * HIDN * 2);
  float*    Z2   = (float*)take((size_t)Bn * OUTC * CIN * 4);

  // --- prep / convert
  k_cvt_pad<<<(HIDN * CINP + 255) / 256, 256, 0, stream>>>(W1, W1h, HIDN, CIN, CINP,
                                                           HIDN * CINP);
  k_cvt_pad<<<(32 * HIDN + 255) / 256, 256, 0, stream>>>(W2, W2h, OUTC, HIDN, HIDN,
                                                         32 * HIDN);
  k_cvt_pad<<<(HIDN * Tn + 255) / 256, 256, 0, stream>>>(Wl1, Wl1h, HIDN, Tn, Tn,
                                                         HIDN * Tn);
  k_cvt_pad<<<(32 * HIDN + 255) / 256, 256, 0, stream>>>(Wl2, Wl2h, OUTC, HIDN, HIDN,
                                                         32 * HIDN);
  k_build_xt<<<(Bn * Tn * CINP) / 256, 256, 0, stream>>>(X, Xt);
  k_build_xp<<<(Bn * CIN * Tn) / 256, 256, 0, stream>>>(X, perm, Xp);

  // --- branch 1
  k_gemm1<<<(Bn * Tn) / 16, 256, 0, stream>>>(W1h, Xt, Y1h);          // 4096 blocks
  k_scan1<<<(Bn * HIDN) / 256, 256, 0, stream>>>(Y1h, S1h);
  k_gemm2<<<(Bn * Tn) / 128, 256, 0, stream>>>(W2h, S1h, Y2);         // 512 blocks
  k_scan2<<<(Bn * OUTC + 255) / 256, 256, 0, stream>>>(Y2, out);

  // --- branch 2
  k_gemm3<<<(Bn * CIN) / 16, 256, 0, stream>>>(Wl1h, Xp, Z1);         // 312 blocks
  k_scan3<<<(Bn * HIDN) / 256, 256, 0, stream>>>(Z1, L1h);
  k_gemm4<<<(Bn * CIN) / 128, 256, 0, stream>>>(Wl2h, L1h, Z2);       // 39 blocks
  k_scan4<<<(Bn * OUTC + 255) / 256, 256, 0, stream>>>(Z2, out);
}
